// GLMAttention_22144851378666
// MI455X (gfx1250) — compile-verified, matrix operand order checked by reading, not run on hardware
//
#include <hip/hip_runtime.h>

typedef __bf16 bf16;
typedef __attribute__((ext_vector_type(16))) __bf16 v16bf;
typedef __attribute__((ext_vector_type(8)))  float  v8f;
typedef __attribute__((ext_vector_type(4)))  __bf16 bf16x4;

#define HIDDEN   4096
#define NQKV     5120
#define TTOK     4096
#define SEQ      1024
#define NHEADS   32
#define NKV      4
#define HDIM     128
#define ATT_SCALE 0.08838834764831845f  // 1/sqrt(128)

#if defined(__has_builtin)
#if __has_builtin(__builtin_amdgcn_global_load_async_to_lds_b128)
#define HAVE_ASYNC_LDS 1
#else
#define HAVE_ASYNC_LDS 0
#endif
#else
#define HAVE_ASYNC_LDS 0
#endif

__device__ __forceinline__ float bf2f(bf16 x) {
  unsigned short h = __builtin_bit_cast(unsigned short, x);
  unsigned u = ((unsigned)h) << 16;
  return __builtin_bit_cast(float, u);
}
__device__ __forceinline__ bf16 f2bf(float f) {
  unsigned u = __builtin_bit_cast(unsigned, f);
  u += 0x7FFFu + ((u >> 16) & 1u);   // round-to-nearest-even
  unsigned short h = (unsigned short)(u >> 16);
  return __builtin_bit_cast(bf16, h);
}

// 16-byte global -> LDS copy; async (ASYNCcnt) when the gfx1250 builtin is
// available, otherwise synchronous through VGPRs.  Both src and dst must be
// 16-byte aligned.
#if HAVE_ASYNC_LDS
typedef int v4i_raw __attribute__((vector_size(16)));
typedef __attribute__((address_space(1))) v4i_raw* gas1_v4i;
typedef __attribute__((address_space(3))) v4i_raw* las3_v4i;
#endif

__device__ __forceinline__ void cp_async16(const bf16* g, bf16* l) {
#if HAVE_ASYNC_LDS
  __builtin_amdgcn_global_load_async_to_lds_b128(
      (gas1_v4i)(uintptr_t)g,
      (las3_v4i)(unsigned)(uintptr_t)l, 0, 0);
#else
  const bf16x4* s = (const bf16x4*)g;
  bf16x4* d = (bf16x4*)l;
  d[0] = s[0];
  d[1] = s[1];
#endif
}
__device__ __forceinline__ void wait_async() {
#if HAVE_ASYNC_LDS
#if __has_builtin(__builtin_amdgcn_s_wait_asynccnt)
  __builtin_amdgcn_s_wait_asynccnt(0);
#else
  asm volatile("s_wait_asynccnt 0x0" ::: "memory");
#endif
#endif
}

// Load a 16x32 bf16 operand fragment (A layout; also used for B when the
// tile is stored transposed [n][k]).  ISA layout for 16-bit A 16x32:
// lanes 0-15 : row = lane,    elements 0..7 -> K 0..7,  8..15 -> K 16..23
// lanes 16-31: row = lane-16, elements 0..7 -> K 8..15, 8..15 -> K 24..31
__device__ __forceinline__ v16bf load_frag(const bf16* base, int stride) {
  int lane = threadIdx.x & 31;
  const bf16* p = base + (lane & 15) * stride + ((lane >> 4) << 3);
  bf16x4 a0 = *(const bf16x4*)(p);
  bf16x4 a1 = *(const bf16x4*)(p + 4);
  bf16x4 a2 = *(const bf16x4*)(p + 16);
  bf16x4 a3 = *(const bf16x4*)(p + 20);
  v16bf r;
  r[0]=a0[0];  r[1]=a0[1];  r[2]=a0[2];  r[3]=a0[3];
  r[4]=a1[0];  r[5]=a1[1];  r[6]=a1[2];  r[7]=a1[3];
  r[8]=a2[0];  r[9]=a2[1];  r[10]=a2[2]; r[11]=a2[3];
  r[12]=a3[0]; r[13]=a3[1]; r[14]=a3[2]; r[15]=a3[3];
  return r;
}

__device__ __forceinline__ v8f wmma_bf16(v16bf a, v16bf b, v8f c) {
  return __builtin_amdgcn_wmma_f32_16x16x32_bf16(false, a, false, b,
                                                 (short)0, c, false, false);
}

__device__ __forceinline__ float redmax16(float v) {
  v = fmaxf(v, __shfl_xor(v, 1, 32));
  v = fmaxf(v, __shfl_xor(v, 2, 32));
  v = fmaxf(v, __shfl_xor(v, 4, 32));
  v = fmaxf(v, __shfl_xor(v, 8, 32));
  return v;
}
__device__ __forceinline__ float redsum16(float v) {
  v += __shfl_xor(v, 1, 32);
  v += __shfl_xor(v, 2, 32);
  v += __shfl_xor(v, 4, 32);
  v += __shfl_xor(v, 8, 32);
  return v;
}

// ---------------------------------------------------------------------------
// Generic 128x128x(K) WMMA GEMM. 256 threads = 8 waves (2M x 4N), each wave
// computes 64x32 via 4x2 16x16 accumulators.  B tile staged transposed with a
// per-thread 4x4 micro-transpose (b64 LDS stores).
// QKV==true : C = A(f32) * B + bias, scattered as bf16 into Q/K/V buffers
// QKV==false: C = A(bf16) * B, written f32 to outF
// ---------------------------------------------------------------------------
template<bool ABF16, bool QKV>
__global__ __launch_bounds__(256) void gemm_wmma_kernel(
    const float* __restrict__ Af, const bf16* __restrict__ Abf,
    const float* __restrict__ B,  const float* __restrict__ bias,
    bf16* __restrict__ outQ, bf16* __restrict__ outK, bf16* __restrict__ outV,
    float* __restrict__ outF, int M, int N, int K)
{
  __shared__ bf16 sA[128 * 40];
  __shared__ bf16 sB[128 * 40];   // transposed: [n][k]

  const int tid  = threadIdx.x;
  const int lane = tid & 31;
  const int w    = tid >> 5;
  const int wm   = (w & 1) * 64;
  const int wn   = (w >> 1) * 32;
  const int n0   = blockIdx.x * 128;
  const int m0   = blockIdx.y * 128;

  v8f acc[4][2];
#pragma unroll
  for (int mi = 0; mi < 4; ++mi)
#pragma unroll
    for (int ni = 0; ni < 2; ++ni)
#pragma unroll
      for (int r = 0; r < 8; ++r) acc[mi][ni][r] = 0.f;

  for (int k0 = 0; k0 < K; k0 += 32) {
    // prefetch the next K-step's global lines (weights are streamed once)
    if (k0 + 32 < K) {
      const char* pfA = ABF16
          ? (const char*)(Abf + (size_t)(m0 + (tid >> 1)) * K + k0 + 32)
          : (const char*)(Af  + (size_t)(m0 + (tid >> 1)) * K + k0 + 32);
      __builtin_prefetch(pfA, 0, 1);
      __builtin_prefetch(B + (size_t)(k0 + 32 + (tid >> 3)) * N + n0 + (tid & 7) * 16, 0, 1);
    }

    // ---- stage A tile (128 x 32) ----
    if (ABF16) {
#pragma unroll
      for (int i = 0; i < 2; ++i) {
        int c   = tid + i * 256;       // 512 chunks of 8 elems (16B)
        int row = c >> 2;
        int kk  = (c & 3) * 8;
        cp_async16(Abf + (size_t)(m0 + row) * K + k0 + kk, sA + row * 40 + kk);
      }
    } else {
#pragma unroll
      for (int i = 0; i < 4; ++i) {
        int c   = tid + i * 256;       // 1024 chunks of 4
        int row = c >> 3;
        int kk  = (c & 7) * 4;
        float4 v = *(const float4*)(Af + (size_t)(m0 + row) * K + k0 + kk);
        bf16x4 h;
        h[0] = f2bf(v.x); h[1] = f2bf(v.y); h[2] = f2bf(v.z); h[3] = f2bf(v.w);
        *(bf16x4*)(sA + row * 40 + kk) = h;
      }
    }
    // ---- stage B tile (32 x 128) transposed into sB[n][k]; 4x4 per thread ----
    {
      int k4 = (tid >> 5) * 4;         // 8 k-groups of 4
      int n4 = (tid & 31) * 4;         // 32 n-groups of 4 (coalesced)
      float4 rr[4];
#pragma unroll
      for (int j = 0; j < 4; ++j)
        rr[j] = *(const float4*)(B + (size_t)(k0 + k4 + j) * N + n0 + n4);
#pragma unroll
      for (int j2 = 0; j2 < 4; ++j2) {
        bf16x4 h;
        h[0] = f2bf(((const float*)&rr[0])[j2]);
        h[1] = f2bf(((const float*)&rr[1])[j2]);
        h[2] = f2bf(((const float*)&rr[2])[j2]);
        h[3] = f2bf(((const float*)&rr[3])[j2]);
        *(bf16x4*)(sB + (n4 + j2) * 40 + k4) = h;   // b64 store
      }
    }
    wait_async();
    __syncthreads();

    v16bf a[4], bb[2];
#pragma unroll
    for (int mi = 0; mi < 4; ++mi) a[mi]  = load_frag(sA + (wm + mi * 16) * 40, 40);
#pragma unroll
    for (int ni = 0; ni < 2; ++ni) bb[ni] = load_frag(sB + (wn + ni * 16) * 40, 40);
#pragma unroll
    for (int mi = 0; mi < 4; ++mi)
#pragma unroll
      for (int ni = 0; ni < 2; ++ni)
        acc[mi][ni] = wmma_bf16(a[mi], bb[ni], acc[mi][ni]);
    __syncthreads();
  }

  // ---- epilogue (C layout: lane 0-15 col, VGPR r -> row r (+8 for hi lanes)) ----
  const int hi = (lane >> 4) << 3;
  const int cl = lane & 15;
#pragma unroll
  for (int mi = 0; mi < 4; ++mi) {
#pragma unroll
    for (int ni = 0; ni < 2; ++ni) {
      int col  = n0 + wn + ni * 16 + cl;
      float bv = QKV ? bias[col] : 0.f;
#pragma unroll
      for (int r = 0; r < 8; ++r) {
        int row  = m0 + wm + mi * 16 + r + hi;
        float vv = acc[mi][ni][r] + bv;
        if (QKV) {
          if (col < NHEADS * HDIM)                    // Q: heads 0..31
            outQ[(size_t)row * 4096 + col] = f2bf(vv);
          else if (col < (NHEADS + NKV) * HDIM)       // K: heads 32..35
            outK[(size_t)row * 512 + (col - 4096)] = f2bf(vv);
          else                                        // V: heads 36..39
            outV[(size_t)row * 512 + (col - 4608)] = f2bf(vv);
        } else {
          outF[(size_t)row * N + col] = vv;
        }
      }
    }
  }
}

// ---------------------------------------------------------------------------
// RoPE, in place on bf16 Q and K. One thread per rotation pair.
// ---------------------------------------------------------------------------
__global__ void rope_kernel(bf16* __restrict__ Qb, bf16* __restrict__ Kb,
                            const int* __restrict__ pids)
{
  int idx = blockIdx.x * blockDim.x + threadIdx.x;
  const int total = TTOK * (NHEADS + NKV) * 32;    // 32 pairs per head
  if (idx >= total) return;
  int i    = idx & 31;
  int rest = idx >> 5;
  int hh   = rest % (NHEADS + NKV);
  int t    = rest / (NHEADS + NKV);

  float pos = (float)pids[t];
  float inv = __powf(10000.0f, -(float)(2 * i) * (1.0f / 64.0f));
  float f   = pos * inv;
  float sn, cs;
  sincosf(f, &sn, &cs);

  bf16* p;
  if (hh < NHEADS) p = Qb + (size_t)t * 4096 + hh * 128 + 2 * i;
  else             p = Kb + (size_t)t * 512 + (hh - NHEADS) * 128 + 2 * i;
  float xe = bf2f(p[0]);
  float xo = bf2f(p[1]);
  p[0] = f2bf(xe * cs - xo * sn);
  p[1] = f2bf(xo * cs + xe * sn);
}

// ---------------------------------------------------------------------------
// Flash attention. Grid (S/128, NHEADS, B); 256 threads = 8 waves; wave w
// owns q rows [bx*128 + 16w, +16).  kv processed in 32-wide chunks.
// ---------------------------------------------------------------------------
__global__ __launch_bounds__(256) void attn_wmma_kernel(
    const bf16* __restrict__ Qb, const bf16* __restrict__ Kb,
    const bf16* __restrict__ Vb, bf16* __restrict__ Ob)
{
  __shared__ bf16 sK[32 * 136];        // [kv][d]   (== Bt for S = Q K^T)
  __shared__ bf16 sV[128 * 40];        // [d][kv]   (== Bt for O = P V)
  __shared__ bf16 sP[8 * 16 * 40];     // per-wave P, [qrow][kv]

  const int tid  = threadIdx.x;
  const int lane = tid & 31;
  const int w    = tid >> 5;
  const int bx   = blockIdx.x;
  const int head = blockIdx.y;
  const int b    = blockIdx.z;
  const int kvh  = head >> 3;          // g = 8
  const int q0   = bx * 128 + w * 16;
  const int hi   = (lane >> 4) << 3;
  const int cl   = lane & 15;

  // preload Q fragments (4 d-chunks of 32) straight from global
  const bf16* qbase = Qb + (size_t)(b * SEQ + q0) * 4096 + head * 128;
  v16bf aQ[4];
#pragma unroll
  for (int kc = 0; kc < 4; ++kc) aQ[kc] = load_frag(qbase + kc * 32, 4096);

  v8f o[8];
  float mrow[8], lrow[8];
#pragma unroll
  for (int d = 0; d < 8; ++d)
#pragma unroll
    for (int r = 0; r < 8; ++r) o[d][r] = 0.f;
#pragma unroll
  for (int r = 0; r < 8; ++r) { mrow[r] = -1e30f; lrow[r] = 0.f; }

  const int iters = (bx + 1) * 4;      // uniform across block -> barriers legal
  for (int it = 0; it < iters; ++it) {
    const int c0 = it * 32;
    // ---- stage K (32 x 128, row major) -- async to LDS when available ----
#pragma unroll
    for (int i = 0; i < 2; ++i) {
      int c   = tid + i * 256;         // 512 chunks of 8 elems (16B)
      int row = c >> 4;
      int ch  = (c & 15) * 8;
      cp_async16(Kb + (size_t)(b * SEQ + c0 + row) * 512 + kvh * 128 + ch,
                 sK + row * 136 + ch);
    }
    // ---- stage V transposed -> sV[d][kv]; 4x4 micro-transpose per thread ----
    {
      int d4  = (tid & 31) * 4;        // 32 d-groups of 4 (coalesced-ish)
      int kv4 = (tid >> 5) * 4;        // 8 kv-groups of 4
      bf16x4 vv[4];
#pragma unroll
      for (int j = 0; j < 4; ++j)
        vv[j] = *(const bf16x4*)(Vb + (size_t)(b * SEQ + c0 + kv4 + j) * 512 +
                                 kvh * 128 + d4);
#pragma unroll
      for (int j2 = 0; j2 < 4; ++j2) {
        bf16x4 h;
        h[0] = vv[0][j2]; h[1] = vv[1][j2]; h[2] = vv[2][j2]; h[3] = vv[3][j2];
        *(bf16x4*)(sV + (d4 + j2) * 40 + kv4) = h;   // b64 store
      }
    }
    wait_async();
    __syncthreads();

    // ---- scores: two 16x16 tiles over kv chunk of 32 ----
    v8f s0, s1;
#pragma unroll
    for (int r = 0; r < 8; ++r) { s0[r] = 0.f; s1[r] = 0.f; }
#pragma unroll
    for (int kc = 0; kc < 4; ++kc) {
      v16bf b0 = load_frag(sK + kc * 32, 136);
      v16bf b1 = load_frag(sK + 16 * 136 + kc * 32, 136);
      s0 = wmma_bf16(aQ[kc], b0, s0);
      s1 = wmma_bf16(aQ[kc], b1, s1);
    }

    // ---- online softmax (row = r + hi across lanes 0-15 of each half) ----
#pragma unroll
    for (int r = 0; r < 8; ++r) {
      int rowg = q0 + r + hi;
      float v0 = s0[r] * ATT_SCALE;
      float v1 = s1[r] * ATT_SCALE;
      if (c0 + cl > rowg)      v0 = -1e30f;
      if (c0 + 16 + cl > rowg) v1 = -1e30f;
      float mx   = redmax16(fmaxf(v0, v1));
      float mnew = fmaxf(mrow[r], mx);
      float sc   = __expf(mrow[r] - mnew);
      float p0   = __expf(v0 - mnew);
      float p1   = __expf(v1 - mnew);
      float rs   = redsum16(p0 + p1);
      lrow[r] = lrow[r] * sc + rs;
      mrow[r] = mnew;
#pragma unroll
      for (int d = 0; d < 8; ++d) o[d][r] *= sc;
      bf16* pw = sP + (w * 16 + r + hi) * 40;
      pw[cl]      = f2bf(p0);
      pw[16 + cl] = f2bf(p1);
    }
    __syncthreads();

    // ---- O += P * V ----
    v16bf aP = load_frag(sP + w * 16 * 40, 40);
#pragma unroll
    for (int d = 0; d < 8; ++d) {
      v16bf bV = load_frag(sV + d * 16 * 40, 40);
      o[d] = wmma_bf16(aP, bV, o[d]);
    }
    __syncthreads();
  }

  // ---- normalize and store bf16 attn output ----
#pragma unroll
  for (int r = 0; r < 8; ++r) {
    float inv = 1.0f / lrow[r];
    int t = b * SEQ + q0 + r + hi;
    bf16* dst = Ob + (size_t)t * 4096 + head * 128;
#pragma unroll
    for (int d = 0; d < 8; ++d) dst[d * 16 + cl] = f2bf(o[d][r] * inv);
  }
}

// ---------------------------------------------------------------------------
extern "C" void kernel_launch(void* const* d_in, const int* in_sizes, int n_in,
                              void* d_out, int out_size, void* d_ws, size_t ws_size,
                              hipStream_t stream) {
  const float* hs      = (const float*)d_in[0];
  const float* w_qkv   = (const float*)d_in[1];
  const float* b_qkv   = (const float*)d_in[2];
  const float* w_dense = (const float*)d_in[3];
  const int*   pids    = (const int*)d_in[4];
  float*       out     = (float*)d_out;

  char* ws = (char*)d_ws;
  bf16* Qb = (bf16*)(ws);                         // 4096*4096*2 = 32 MiB
  bf16* Kb = (bf16*)(ws + 33554432ull);           // 4096*512*2  =  4 MiB
  bf16* Vb = (bf16*)(ws + 37748736ull);           // 4096*512*2  =  4 MiB
  bf16* Ab = (bf16*)(ws + 41943040ull);           // 4096*4096*2 = 32 MiB

  // 1) QKV projection + bias, scatter bf16 Q/K/V
  gemm_wmma_kernel<false, true><<<dim3(NQKV / 128, TTOK / 128), 256, 0, stream>>>(
      hs, nullptr, w_qkv, b_qkv, Qb, Kb, Vb, nullptr, TTOK, NQKV, HIDDEN);

  // 2) RoPE in place on Q and K
  {
    int total = TTOK * (NHEADS + NKV) * 32;
    rope_kernel<<<(total + 255) / 256, 256, 0, stream>>>(Qb, Kb, pids);
  }

  // 3) causal GQA flash attention
  attn_wmma_kernel<<<dim3(SEQ / 128, NHEADS, 4), 256, 0, stream>>>(Qb, Kb, Vb, Ab);

  // 4) output projection, f32 result
  gemm_wmma_kernel<true, false><<<dim3(HIDDEN / 128, TTOK / 128), 256, 0, stream>>>(
      nullptr, Ab, w_dense, nullptr, nullptr, nullptr, nullptr, out, TTOK, HIDDEN, HIDDEN);
}
